// GatedDeltaNet_90357521973986
// MI455X (gfx1250) — matrix operand order repaired and checked by reading, not compiled, not run on hardware
//
#include <hip/hip_runtime.h>

#define Tt 1024
#define Hh 2048
#define HKc 8
#define HVc 16
#define DKc 128
#define DVc 128
#define KDc 1024
#define VDc 2048
#define CDc 4096
#define NQKVZ 6144
#define EPSf 1e-6f
#define CH 32

typedef __attribute__((ext_vector_type(16))) __bf16 v16bf;
typedef __attribute__((ext_vector_type(8)))  float  v8f;

// ---------------------------------------------------------------------------
// C[M,N] = A[M,K] * B[K,N], fp32 in/out, bf16 WMMA compute (f32 accumulate).
// Block: 128 threads = 4 wave32. Tile: BM=64 (16 rows/wave), BN=64, BK=64.
// Software-pipelined + double-buffered LDS: global loads for chunk i+1 are
// issued BEFORE the 8 WMMAs of chunk i; converts/LDS-stores (the consumers
// that force s_wait_loadcnt) run AFTER the WMMAs. One barrier per chunk.
// Requires M%64==0, N%64==0, K%64==0 (true for all call sites here).
// ---------------------------------------------------------------------------
__global__ __launch_bounds__(128)
void gemm_bf16_wmma(const float* __restrict__ A, const float* __restrict__ B,
                    float* __restrict__ C, int M, int N, int Kd) {
    // 72-element row stride = 144 B = multiple of 16 B -> b128-friendly
    __shared__ __bf16 As[2][64][72];
    __shared__ __bf16 Bs[2][64][72];

    const int tid  = threadIdx.x;
    const int lane = tid & 31;
    const int wv   = tid >> 5;      // wave id 0..3 -> 16-row M slice
    const int half = lane >> 4;     // 0 | 1
    const int l16  = lane & 15;
    const int m0   = blockIdx.y * 64;
    const int n0   = blockIdx.x * 64;
    const int mrow = m0 + wv * 16;

    v8f acc[4];
#pragma unroll
    for (int i = 0; i < 4; ++i) acc[i] = (v8f){0.f,0.f,0.f,0.f,0.f,0.f,0.f,0.f};

    float4 areg[8];      // staged A quads (row-major, K contiguous)
    float2 breg[16];     // staged B k-pairs per column

    // issue global loads for chunk at k0 into registers (no consumers here)
    auto loadreg = [&](int k0) {
#pragma unroll
        for (int ii = 0; ii < 8; ++ii) {
            int q  = tid + ii * 128;
            int r  = q >> 4;                 // row 0..63
            int kq = (q & 15) << 2;          // k offset 0,4,...,60
            areg[ii] = *(const float4*)&A[(size_t)(m0 + r) * Kd + k0 + kq];
        }
#pragma unroll
        for (int ii = 0; ii < 16; ++ii) {
            int p  = tid + ii * 128;         // 0..2047
            int n  = p & 63;
            int kp = p >> 6;                 // kk = 2*kp, 2*kp+1
            breg[ii].x = B[(size_t)(k0 + 2 * kp)     * N + n0 + n];
            breg[ii].y = B[(size_t)(k0 + 2 * kp + 1) * N + n0 + n];
        }
    };
    // convert staged registers to bf16 and store into LDS buffer `buf`
    auto storeLDS = [&](int buf) {
#pragma unroll
        for (int ii = 0; ii < 8; ++ii) {
            int q  = tid + ii * 128;
            int r  = q >> 4;
            int kq = (q & 15) << 2;
            union { __bf16 h[4]; uint2 u; } t;
            t.h[0] = (__bf16)areg[ii].x; t.h[1] = (__bf16)areg[ii].y;
            t.h[2] = (__bf16)areg[ii].z; t.h[3] = (__bf16)areg[ii].w;
            *(uint2*)&As[buf][r][kq] = t.u;  // 8B-aligned b64 store
        }
#pragma unroll
        for (int ii = 0; ii < 16; ++ii) {
            int p  = tid + ii * 128;
            int n  = p & 63;
            int kp = p >> 6;
            union { __bf16 h[2]; unsigned u; } t;
            t.h[0] = (__bf16)breg[ii].x; t.h[1] = (__bf16)breg[ii].y;
            *(unsigned*)&Bs[buf][n][2 * kp] = t.u;
        }
    };

    loadreg(0);
    storeLDS(0);
    const int nchunks = Kd >> 6;
    for (int ci = 0; ci < nchunks; ++ci) {
        __syncthreads();                       // buffer p fully staged
        const int p = ci & 1;
        const bool more = (ci + 1 < nchunks);
        if (more) loadreg((ci + 1) << 6);      // in-flight during WMMAs below
        if (ci + 2 < nchunks) {
            __builtin_prefetch(&A[(size_t)(m0 + (tid >> 1)) * Kd + ((ci + 2) << 6)], 0, 1);
            __builtin_prefetch(&B[(size_t)(((ci + 2) << 6) + (tid >> 1)) * N + n0], 0, 1);
        }
#pragma unroll
        for (int ks = 0; ks < 64; ks += 32) {
            // A fragment (ISA layout): row = l16, runs [8h,+8) and [16+8h,+8)
            const __bf16* arow = &As[p][wv * 16 + l16][ks];
            union { uint4 u[2]; v16bf v; } ta;
            ta.u[0] = *(const uint4*)(arow + 8 * half);
            ta.u[1] = *(const uint4*)(arow + 16 + 8 * half);
            v16bf af = ta.v;
            // preload all 4 B fragments, then chain WMMAs
            v16bf bfrag[4];
#pragma unroll
            for (int nt = 0; nt < 4; ++nt) {
                const __bf16* bp = &Bs[p][nt * 16 + l16][ks] + 8 * half;
                union { uint4 u[2]; v16bf v; } tb;
                tb.u[0] = *(const uint4*)(bp);
                tb.u[1] = *(const uint4*)(bp + 16);
                bfrag[nt] = tb.v;
            }
#pragma unroll
            for (int nt = 0; nt < 4; ++nt)
                acc[nt] = __builtin_amdgcn_wmma_f32_16x16x32_bf16(
                    false, af, false, bfrag[nt], (short)0, acc[nt], false, false);
        }
        if (more) storeLDS(p ^ 1);             // waits for loads land here
    }

    // C/D layout: VGPR r -> row r + 8*half, col = l16
#pragma unroll
    for (int nt = 0; nt < 4; ++nt)
#pragma unroll
        for (int r = 0; r < 8; ++r)
            C[(size_t)(mrow + r + 8 * half) * N + n0 + nt * 16 + l16] = acc[nt][r];
}

// ---------------------------------------------------------------------------
// ba = x @ w_ba  (1024x2048 * 2048x32)
// ---------------------------------------------------------------------------
__global__ __launch_bounds__(32)
void ba_kernel(const float* __restrict__ x, const float* __restrict__ w_ba,
               float* __restrict__ ba) {
    int t = blockIdx.x, c = threadIdx.x;
    float s = 0.f;
#pragma unroll 4
    for (int j = 0; j < Hh; ++j) s = fmaf(x[(size_t)t * Hh + j], w_ba[j * 32 + c], s);
    ba[t * 32 + c] = s;
}

// ---------------------------------------------------------------------------
// causal depthwise conv1d (K=4) over the first 4096 channels of qkvz, + SiLU
// ---------------------------------------------------------------------------
__global__ __launch_bounds__(256)
void conv_silu_kernel(const float* __restrict__ qkvz, const float* __restrict__ conv_w,
                      const float* __restrict__ conv_b, float* __restrict__ cv) {
    int t = blockIdx.x;
#pragma unroll
    for (int ii = 0; ii < CDc / 256; ++ii) {
        int c = threadIdx.x + ii * 256;
        float acc = conv_b[c];
#pragma unroll
        for (int j = 0; j < 4; ++j) {
            int ts = t - 3 + j;
            if (ts >= 0) acc = fmaf(qkvz[(size_t)ts * NQKVZ + c], conv_w[c * 4 + j], acc);
        }
        cv[(size_t)t * CDc + c] = acc / (1.f + __expf(-acc));   // silu
    }
}

// ---------------------------------------------------------------------------
// per-timestep prep: l2norm(q)*DK^-0.5, l2norm(k), beta=sigmoid(b),
// g = -exp(a_log)*softplus(a+dt_bias)
// ---------------------------------------------------------------------------
__global__ __launch_bounds__(128)
void prep_kernel(const float* __restrict__ cv, const float* __restrict__ ba,
                 const float* __restrict__ a_log, const float* __restrict__ dt_bias,
                 float* __restrict__ qn, float* __restrict__ kn,
                 float* __restrict__ g, float* __restrict__ beta) {
    int t = blockIdx.x, d = threadIdx.x;
    __shared__ float rq[128], rk[128];
    for (int h = 0; h < HKc; ++h) {
        float qv = cv[(size_t)t * CDc + h * DKc + d];
        float kv = cv[(size_t)t * CDc + KDc + h * DKc + d];
        rq[d] = qv * qv; rk[d] = kv * kv;
        __syncthreads();
#pragma unroll
        for (int s = 64; s > 0; s >>= 1) {
            if (d < s) { rq[d] += rq[d + s]; rk[d] += rk[d + s]; }
            __syncthreads();
        }
        float qi = rsqrtf(rq[0] + EPSf) * 0.08838834764831845f;  // 1/sqrt(128)
        float ki = rsqrtf(rk[0] + EPSf);
        __syncthreads();
        qn[(size_t)t * KDc + h * DKc + d] = qv * qi;
        kn[(size_t)t * KDc + h * DKc + d] = kv * ki;
    }
    if (d < HVc) {
        float bv = ba[t * 32 + d];
        float av = ba[t * 32 + HVc + d];
        beta[t * HVc + d] = 1.f / (1.f + __expf(-bv));
        float xx = av + dt_bias[d];
        float sp = (xx > 20.f) ? xx : log1pf(__expf(xx));
        g[t * HVc + d] = -__expf(a_log[d]) * sp;
    }
}

// ---------------------------------------------------------------------------
// gated delta rule recurrence. One block per value head; thread = dv column,
// S[:,dv] (128 f32) lives in registers. Chunk staging uses CDNA5 async
// global->LDS DMA (ASYNCcnt) -- no VGPR round trip for q/k/v tiles.
// ---------------------------------------------------------------------------
__global__ __launch_bounds__(128)
void recur_kernel(const float* __restrict__ qn, const float* __restrict__ kn,
                  const float* __restrict__ cv, const float* __restrict__ g,
                  const float* __restrict__ beta, float* __restrict__ o) {
    const int hv = blockIdx.x;
    const int kh = hv >> 1;            // GQA: groups = HV/HK = 2
    const int d  = threadIdx.x;        // dv
    __shared__ float qs[CH][DKc], ks[CH][DKc], vs[CH][DVc];
    __shared__ float gs[CH], bs[CH];

    float S[DKc];
#pragma unroll
    for (int i = 0; i < DKc; ++i) S[i] = 0.f;

    for (int c0 = 0; c0 < Tt; c0 += CH) {
        // async global->LDS staging: 1024 float4 slots per array, 8 per thread
#pragma unroll
        for (int ii = 0; ii < 8; ++ii) {
            int s  = d + ii * 128;
            int tt = s >> 5;                 // row in chunk
            int e4 = (s & 31) << 2;          // float offset in row
            unsigned long long gq =
                (unsigned long long)&qn[(size_t)(c0 + tt) * KDc + kh * DKc + e4];
            unsigned long long gk =
                (unsigned long long)&kn[(size_t)(c0 + tt) * KDc + kh * DKc + e4];
            unsigned long long gv =
                (unsigned long long)&cv[(size_t)(c0 + tt) * CDc + 2 * KDc + hv * DVc + e4];
            unsigned lq = (unsigned)(size_t)&qs[tt][e4];
            unsigned lk = (unsigned)(size_t)&ks[tt][e4];
            unsigned lv = (unsigned)(size_t)&vs[tt][e4];
            asm volatile("global_load_async_to_lds_b128 %0, %1, off"
                         :: "v"(lq), "v"(gq) : "memory");
            asm volatile("global_load_async_to_lds_b128 %0, %1, off"
                         :: "v"(lk), "v"(gk) : "memory");
            asm volatile("global_load_async_to_lds_b128 %0, %1, off"
                         :: "v"(lv), "v"(gv) : "memory");
        }
        if (d < CH) {
            gs[d] = g[(c0 + d) * HVc + hv];
            bs[d] = beta[(c0 + d) * HVc + hv];
        }
        asm volatile("s_wait_asynccnt 0x0" ::: "memory");
        __syncthreads();
        for (int tt = 0; tt < CH; ++tt) {
            float eg = __expf(gs[tt]);
            float pred = 0.f;
#pragma unroll
            for (int dk = 0; dk < DKc; ++dk) pred = fmaf(ks[tt][dk], S[dk], pred);
            pred *= eg;                                    // decay applied to S
            float delta = (vs[tt][d] - pred) * bs[tt];
            float out = 0.f;
#pragma unroll
            for (int dk = 0; dk < DKc; ++dk) {
                S[dk] = fmaf(eg, S[dk], ks[tt][dk] * delta);
                out = fmaf(qs[tt][dk], S[dk], out);
            }
            o[(size_t)(c0 + tt) * VDc + hv * DVc + d] = out;
        }
        __syncthreads();
    }
}

// ---------------------------------------------------------------------------
// gated RMSNorm: h = o * silu(z); h *= rsqrt(mean(h^2)+eps); h *= norm_w
// ---------------------------------------------------------------------------
__global__ __launch_bounds__(128)
void gnorm_kernel(const float* __restrict__ o, const float* __restrict__ qkvz,
                  const float* __restrict__ norm_w, float* __restrict__ hbuf) {
    int t = blockIdx.x, hv = blockIdx.y, d = threadIdx.x;
    __shared__ float red[128];
    float ov = o[(size_t)t * VDc + hv * DVc + d];
    float zv = qkvz[(size_t)t * NQKVZ + 2 * KDc + VDc + hv * DVc + d];
    float hval = ov * (zv / (1.f + __expf(-zv)));
    red[d] = hval * hval;
    __syncthreads();
#pragma unroll
    for (int s = 64; s > 0; s >>= 1) {
        if (d < s) red[d] += red[d + s];
        __syncthreads();
    }
    float r = rsqrtf(red[0] * (1.f / (float)DVc) + EPSf);
    hbuf[(size_t)t * VDc + hv * DVc + d] = hval * r * norm_w[d];
}

// ---------------------------------------------------------------------------
extern "C" void kernel_launch(void* const* d_in, const int* in_sizes, int n_in,
                              void* d_out, int out_size, void* d_ws, size_t ws_size,
                              hipStream_t stream) {
    const float* x       = (const float*)d_in[0];
    const float* w_qkvz  = (const float*)d_in[1];
    const float* w_ba    = (const float*)d_in[2];
    const float* conv_w  = (const float*)d_in[3];
    const float* conv_b  = (const float*)d_in[4];
    const float* a_log   = (const float*)d_in[5];
    const float* dt_bias = (const float*)d_in[6];
    const float* norm_w  = (const float*)d_in[7];
    const float* w_o     = (const float*)d_in[8];
    float* out = (float*)d_out;

    float* ws   = (float*)d_ws;
    float* qkvz = ws;                                   // T * 6144
    float* ba   = qkvz + (size_t)Tt * NQKVZ;            // T * 32
    float* cv   = ba   + (size_t)Tt * 32;               // T * 4096
    float* qn   = cv   + (size_t)Tt * CDc;              // T * 1024
    float* kn   = qn   + (size_t)Tt * KDc;              // T * 1024
    float* g    = kn   + (size_t)Tt * KDc;              // T * 16
    float* beta = g    + (size_t)Tt * HVc;              // T * 16
    float* o    = beta + (size_t)Tt * HVc;              // T * 2048
    float* hbuf = o    + (size_t)Tt * VDc;              // T * 2048

    gemm_bf16_wmma<<<dim3(NQKVZ / 64, Tt / 64), 128, 0, stream>>>(
        x, w_qkvz, qkvz, Tt, NQKVZ, Hh);
    ba_kernel<<<Tt, 32, 0, stream>>>(x, w_ba, ba);
    conv_silu_kernel<<<Tt, 256, 0, stream>>>(qkvz, conv_w, conv_b, cv);
    prep_kernel<<<Tt, 128, 0, stream>>>(cv, ba, a_log, dt_bias, qn, kn, g, beta);
    recur_kernel<<<HVc, 128, 0, stream>>>(qn, kn, cv, g, beta, o);
    gnorm_kernel<<<dim3(Tt, HVc), 128, 0, stream>>>(o, qkvz, norm_w, hbuf);
    gemm_bf16_wmma<<<dim3(Hh / 64, Tt / 64), 128, 0, stream>>>(
        hbuf, w_o, out, Tt, Hh, VDc);
}
